// SetConvGridEncoder_13417477833599
// MI455X (gfx1250) — compile-verified, matrix operand order checked by reading, not run on hardware
//
#include <hip/hip_runtime.h>
#include <hip/hip_bf16.h>

// Problem constants (from reference)
#define BATCH 4
#define NPTS  4096      // off-grid points (K dimension)
#define MROWS 4096      // 64x64 grid points
#define DZ    128
#define NCHUNKS (NPTS / 32)
#define ZSTRIDE 40      // f16 elements per LDS row (32 data + 8 pad); 80B, 16B-aligned

typedef __attribute__((ext_vector_type(16))) _Float16 v16h;
typedef __attribute__((ext_vector_type(8)))  _Float16 v8h;
typedef __attribute__((ext_vector_type(8)))  float    v8f;

__global__ __launch_bounds__(128)
void setconv_grid_encoder_kernel(const float* __restrict__ xc_off,   // [B,N,2]
                                 const float* __restrict__ zc_off,   // [B,N,DZ]
                                 const float* __restrict__ zc_on,    // [B,M,DZ]
                                 const float* __restrict__ lsp,      // [2]
                                 const int*   __restrict__ ignore_flag,
                                 float* __restrict__ out)            // [B,M,DZ]
{
    // Double-buffered LDS: zc_off chunk transposed to [dz][k] f16
    // (80B row stride: 16B-aligned b128 reads, 20-bank stride -> conflict-free),
    // plus the 32 xc_off points of each chunk.
    __shared__ __align__(16) _Float16 zT[2][DZ * ZSTRIDE];   // 2 x 10240 B
    __shared__ __align__(16) float    xcT[2][64];            // 2 x 256 B

    const int tid  = threadIdx.x;          // [0,128)
    const int lane = tid & 31;             // wave32
    const int wave = tid >> 5;             // 4 waves / block
    const int bb   = blockIdx.x;           // 256 blocks
    const int b    = bb >> 6;              // batch
    const int rowBase = (bb & 63) << 6;    // 64 grid rows per block

    // lengthscale: 1e-5 + softplus(p); fold inverse into grid coords once
    const float inv0 = 1.0f / (1e-5f + log1pf(__expf(lsp[0])));
    const float inv1 = 1.0f / (1e-5f + log1pf(__expf(lsp[1])));

    // A-fragment geometry (16-bit A 16x32 layout):
    //   lane%16 = M row, lane/16 selects K half; element e -> K = e + 8h + (e&8)
    const int h  = lane >> 4;
    const int ml = lane & 15;
    const int mg = rowBase + wave * 16 + ml;            // this lane's A row (global m)
    const float sgy = (float)(mg >> 6) * (1.0f / 63.0f) * inv0;  // meshgrid: row/63
    const float sgx = (float)(mg & 63) * (1.0f / 63.0f) * inv1;  // meshgrid: col/63

    const float* zsrc_b  = zc_off + (size_t)b * NPTS * DZ;
    const float* xcsrc_b = xc_off + (size_t)b * NPTS * 2;

    // ---- prologue: stage chunk 0 into registers ----
    float4 r[8];
    float  xreg = 0.0f;
    #pragma unroll
    for (int rep = 0; rep < 8; ++rep)
        r[rep] = *(const float4*)(zsrc_b + rep * 512 + tid * 4);
    if (tid < 64) xreg = xcsrc_b[tid];

    v8f acc[8] = {};   // 16(m) x 128(dz) strip per wave

    for (int it = 0; it < NCHUNKS; ++it) {
        const int buf = it & 1;

        // ---- drain staged registers -> LDS (f32->f16, transposed) ----
        {
            _Float16* zdst = &zT[buf][0];
            #pragma unroll
            for (int rep = 0; rep < 8; ++rep) {
                const int idx = rep * 512 + tid * 4;
                const int k = idx >> 7;                 // k within chunk
                const int n = idx & 127;                // dz channel
                zdst[(n + 0) * ZSTRIDE + k] = (_Float16)r[rep].x;
                zdst[(n + 1) * ZSTRIDE + k] = (_Float16)r[rep].y;
                zdst[(n + 2) * ZSTRIDE + k] = (_Float16)r[rep].z;
                zdst[(n + 3) * ZSTRIDE + k] = (_Float16)r[rep].w;
            }
            if (tid < 64) xcT[buf][tid] = xreg;
        }
        __syncthreads();   // single barrier per chunk (double buffering)

        // ---- issue next chunk's global loads NOW; they complete behind the
        //      WMMA/exp work below and are only waited on at the next drain ----
        if (it + 1 < NCHUNKS) {
            const float* zsrc = zsrc_b + (size_t)(it + 1) * 32 * DZ;
            #pragma unroll
            for (int rep = 0; rep < 8; ++rep)
                r[rep] = *(const float4*)(zsrc + rep * 512 + tid * 4);
            if (tid < 64) xreg = xcsrc_b[(it + 1) * 64 + tid];
            // pull chunk it+2 toward L2/L0 (global_prefetch_b8)
            if (it + 2 < NCHUNKS)
                __builtin_prefetch(zsrc + 32 * DZ + tid * 32, 0, 1);
        }

        // ---- build A fragment: fused weight computation (never hits memory) ----
        v16h A;
        const float2* xp = (const float2*)&xcT[buf][0];
        #pragma unroll
        for (int e = 0; e < 16; ++e) {
            const int kk = e + h * 8 + (e & 8);         // ISA 16-bit A K-striping
            const float2 pc = xp[kk];
            const float d0 = fmaf(-pc.x, inv0, sgy);    // (xg - xc) / ls
            const float d1 = fmaf(-pc.y, inv1, sgx);
            const float s  = fmaf(d0, d0, d1 * d1);
            A[e] = (_Float16)__expf(-0.5f * s);
        }

        // ---- 8 dz-tiles: B fragment from LDS (2x ds_load_b128) + WMMA ----
        const _Float16* zb = &zT[buf][0];
        #pragma unroll
        for (int j = 0; j < 8; ++j) {
            const int rowd = j * 16 + ml;               // lane's B column (dz index)
            const v8h lo = *(const v8h*)&zb[rowd * ZSTRIDE + h * 8];        // K=8h..8h+7
            const v8h hi = *(const v8h*)&zb[rowd * ZSTRIDE + 16 + h * 8];   // K=16+8h..
            const v16h Bf = __builtin_shufflevector(lo, hi,
                0, 1, 2, 3, 4, 5, 6, 7, 8, 9, 10, 11, 12, 13, 14, 15);
            acc[j] = __builtin_amdgcn_wmma_f32_16x16x32_f16(
                false, A, false, Bf, (short)0, acc[j], false, false);
        }
    }

    // ---- epilogue: D layout VGPR v -> M = v + 8h, N = lane%16 ----
    const int ignore = *ignore_flag;
    #pragma unroll
    for (int j = 0; j < 8; ++j) {
        #pragma unroll
        for (int v = 0; v < 8; ++v) {
            const int mrow = rowBase + wave * 16 + v + 8 * h;
            const size_t o = (((size_t)b * MROWS + mrow) << 7) + j * 16 + ml;
            float rr = acc[j][v];
            if (!ignore) rr += zc_on[o];
            out[o] = rr;
        }
    }
}

extern "C" void kernel_launch(void* const* d_in, const int* in_sizes, int n_in,
                              void* d_out, int out_size, void* d_ws, size_t ws_size,
                              hipStream_t stream) {
    const float* xc_off = (const float*)d_in[0];   // [B,N,2]
    // d_in[1] = xc_on_grid: analytic meshgrid, recomputed on device (never loaded)
    const float* zc_off = (const float*)d_in[2];   // [B,N,DZ]
    const float* zc_on  = (const float*)d_in[3];   // [B,H,W,DZ]
    const float* lsp    = (const float*)d_in[4];   // [2]
    const int*   ign    = (const int*)d_in[5];     // scalar
    float* out = (float*)d_out;                    // [B,H,W,DZ]

    // 4 batches x 64 row-groups; 128 threads = 4 wave32, each wave owns 16x128 output
    setconv_grid_encoder_kernel<<<dim3(BATCH * 64), dim3(128), 0, stream>>>(
        xc_off, zc_off, zc_on, lsp, ign, out);
}
